// _mean_max_sop_22814866277113
// MI455X (gfx1250) — compile-verified, matrix operand order checked by reading, not run on hardware
//
#include <hip/hip_runtime.h>
#include <hip/hip_bf16.h>

typedef __attribute__((ext_vector_type(2)))  float    v2f;
typedef __attribute__((ext_vector_type(8)))  float    v8f;
typedef __attribute__((ext_vector_type(16))) _Float16 v16h;
typedef int i32x4 __attribute__((vector_size(16)));   // async builtin's expected pointee

#if __has_builtin(__builtin_amdgcn_wmma_f32_16x16x4_f32)
#define SOP_HAVE_WMMA_F32K4 1
#else
#define SOP_HAVE_WMMA_F32K4 0
#endif

#if __has_builtin(__builtin_amdgcn_global_load_async_to_lds_b128)
#define SOP_HAVE_ASYNC 1
#else
#define SOP_HAVE_ASYNC 0
#endif

static constexpr int kB       = 8;
static constexpr int kN       = 512;
static constexpr int kD       = 128;
static constexpr int kThreads = 64;                  // 2 waves (wave32)
static constexpr int kLIFloats = kN * 16;            // 32 KB: I-tile columns
static constexpr int kLJFloats = kN * 32;            // 64 KB: J-pair columns
static constexpr int kSmemFloats = kLIFloats + kLJFloats;   // 96 KB total (LDS 320 KB/WGP)

// One 16x16 outer-product tile: D[i][j] = xi[i] * xj[j], via rank-1 WMMA (k=0 only).
__device__ __forceinline__ v8f outer16x16(float xi, float xj, bool act) {
#if SOP_HAVE_WMMA_F32K4
  v2f a = {0.f, 0.f};
  v2f b = {0.f, 0.f};
  a.x = act ? xi : 0.f;
  b.x = act ? xj : 0.f;
  v8f c = {};
  return __builtin_amdgcn_wmma_f32_16x16x4_f32(false, a, false, b, (short)0, c, false, false);
#else
  v16h a = {};
  v16h b = {};
  a[0] = act ? (_Float16)xi : (_Float16)0.f;
  b[0] = act ? (_Float16)xj : (_Float16)0.f;
  v8f c = {};
  return __builtin_amdgcn_wmma_f32_16x16x32_f16(false, a, false, b, (short)0, c, false, false);
#endif
}

// Grid: 256 blocks = B(8) * I-tiles(8) * J-pairs(4). Block = 2 waves; wave w owns
// J-tile (pair*2 + w). LDS stages only the 16 I-cols (32 KB) + 32 J-cols (64 KB)
// this block needs, loaded via async-to-LDS, then read twice (max pass + exp pass).
__global__ __launch_bounds__(kThreads, 1)
void sop_softmax_pool_kernel(const float* __restrict__ x, float* __restrict__ out) {
  extern __shared__ float smem[];        // [kLIFloats | kLJFloats]
  float* __restrict__ LI = smem;
  float* __restrict__ LJ = smem + kLIFloats;

  const int bid  = blockIdx.x;
  const int b    = bid >> 5;             // 8 it * 4 jp = 32 blocks per batch
  const int it   = (bid >> 2) & 7;
  const int jp   = bid & 3;
  const int tid  = threadIdx.x;
  const int lane = tid & 31;
  const int wave = tid >> 5;

  const int I0    = it * 16;
  const int Jbase = jp * 32;
  const int J0    = Jbase + wave * 16;

  const float* xb = x + (size_t)b * kN * kD;

  __builtin_prefetch(xb + I0 + tid, 0, 0);     // global_prefetch_b8

  // ---- stage columns into LDS (16-byte chunks, all 16B-aligned) ----
#if SOP_HAVE_ASYNC
  for (int i = tid; i < kLIFloats / 4; i += kThreads) {    // 2048 chunks
    const int n = i >> 2, c = (i & 3) * 4;
    __builtin_amdgcn_global_load_async_to_lds_b128(
        (i32x4*)(xb + n * kD + I0 + c), (i32x4*)(LI + i * 4), 0, 0);
  }
  for (int i = tid; i < kLJFloats / 4; i += kThreads) {    // 4096 chunks
    const int n = i >> 3, c = (i & 7) * 4;
    __builtin_amdgcn_global_load_async_to_lds_b128(
        (i32x4*)(xb + n * kD + Jbase + c), (i32x4*)(LJ + i * 4), 0, 0);
  }
#if __has_builtin(__builtin_amdgcn_s_wait_asynccnt)
  __builtin_amdgcn_s_wait_asynccnt(0);
#else
  asm volatile("s_wait_asynccnt 0" ::: "memory");
#endif
#else
  for (int i = tid; i < kLIFloats / 4; i += kThreads) {
    const int n = i >> 2, c = (i & 3) * 4;
    ((float4*)LI)[i] = *(const float4*)(xb + n * kD + I0 + c);
  }
  for (int i = tid; i < kLJFloats / 4; i += kThreads) {
    const int n = i >> 3, c = (i & 7) * 4;
    ((float4*)LJ)[i] = *(const float4*)(xb + n * kD + Jbase + c);
  }
#endif
  __syncthreads();

  const int  lm  = lane & 15;
  const bool act = lane < 16;
  const int  liw = lm;                 // LI row offset for this lane
  const int  ljw = wave * 16 + lm;     // LJ row offset for this lane's J-tile

  // ---- pass 1: running max of v_n per (i,j) element ----
  v8f m;
#pragma unroll
  for (int k = 0; k < 8; ++k) m[k] = -3.0e38f;

  for (int n = 0; n < kN; ++n) {
    const float xi = LI[n * 16 + liw];
    const float xj = LJ[n * 32 + ljw];
    const v8f  d  = outer16x16(xi, xj, act);
#pragma unroll
    for (int k = 0; k < 8; ++k) m[k] = fmaxf(m[k], d[k]);
  }

  // ---- pass 2: s1 = sum exp(v-m), s2 = sum exp(v-m)*v ----
  // exp(v-m) = exp2(v*log2e - m*log2e); hoist nm = -m*log2e out of the loop.
  constexpr float kLog2e = 1.4426950408889634f;
  v8f nm;
#pragma unroll
  for (int k = 0; k < 8; ++k) nm[k] = -m[k] * kLog2e;

  v8f s1 = {};
  v8f s2 = {};
  for (int n = 0; n < kN; ++n) {
    const float xi = LI[n * 16 + liw];
    const float xj = LJ[n * 32 + ljw];
    const v8f  d  = outer16x16(xi, xj, act);
#pragma unroll
    for (int k = 0; k < 8; ++k) {
      const float e = __builtin_amdgcn_exp2f(fmaf(d[k], kLog2e, nm[k]));  // v_fma + v_exp
      s1[k] += e;
      s2[k] = fmaf(e, d[k], s2[k]);
    }
  }

  // ---- write out: 16x16 f32 C/D layout: row i = k + 8*(lane>=16), col j = lane&15 ----
  float* ob = out + (size_t)b * kD * kD;
  const int ibase = (lane >> 4) * 8;
#pragma unroll
  for (int k = 0; k < 8; ++k) {
    const int i = ibase + k;
    ob[(size_t)(I0 + i) * kD + (J0 + lm)] = s2[k] / s1[k];
  }
}

extern "C" void kernel_launch(void* const* d_in, const int* in_sizes, int n_in,
                              void* d_out, int out_size, void* d_ws, size_t ws_size,
                              hipStream_t stream) {
  (void)in_sizes; (void)n_in; (void)out_size; (void)d_ws; (void)ws_size;
  const float* x   = (const float*)d_in[0];   // [8, 512, 128] f32
  // d_in[1] (count) and d_in[2] (mask) are unused, matching the reference.
  float* out = (float*)d_out;                 // [8, 16384] f32

  const dim3 grid(kB * 8 * 4);    // batch * I-tiles * J-pairs = 256 blocks
  const dim3 block(kThreads);     // 2 waves
  const size_t shmem = (size_t)kSmemFloats * sizeof(float);  // 96 KB dynamic LDS
  sop_softmax_pool_kernel<<<grid, block, shmem, stream>>>(x, out);
}
